// GNNSoftMask_31756988186744
// MI455X (gfx1250) — matrix operand order repaired and checked by reading, not compiled, not run on hardware
//
#include <hip/hip_runtime.h>
#include <hip/hip_bf16.h>

typedef __attribute__((ext_vector_type(16))) __bf16 v16bf;
typedef __attribute__((ext_vector_type(8)))  __bf16 v8bf;
typedef __attribute__((ext_vector_type(8)))  float  v8f;

#define N_NODES 50000
#define N_EDGES 600000
#define HID     128

static __device__ __forceinline__ v16bf cat8(v8bf lo, v8bf hi) {
  return __builtin_shufflevector(lo, hi, 0,1,2,3,4,5,6,7,8,9,10,11,12,13,14,15);
}

static __device__ __forceinline__ v8bf pack8(float4 f0, float4 f1) {
  v8bf b;
  b[0]=(__bf16)f0.x; b[1]=(__bf16)f0.y; b[2]=(__bf16)f0.z; b[3]=(__bf16)f0.w;
  b[4]=(__bf16)f1.x; b[5]=(__bf16)f1.y; b[6]=(__bf16)f1.z; b[7]=(__bf16)f1.w;
  return b;
}

// ---------------------------------------------------------------------------
// Prep: W[k][n] (f32) -> WT[n][k] (bf16) so WMMA B fragments are contiguous.
// ---------------------------------------------------------------------------
__global__ void transpose_to_bf16(const float* __restrict__ W, __bf16* __restrict__ WT,
                                  int K, int Nc) {
  int idx = blockIdx.x * 256 + threadIdx.x;
  if (idx >= K * Nc) return;
  int k = idx / Nc, n = idx - k * Nc;
  WT[(size_t)n * K + k] = (__bf16)W[idx];
}

__global__ void zero_kernel(float* __restrict__ p, int n) {
  int i = (blockIdx.x * 256 + threadIdx.x) * 4;
  if (i < n) { float4 z = {0.f, 0.f, 0.f, 0.f}; *(float4*)(p + i) = z; }
}

// ---------------------------------------------------------------------------
// out = relu((A0 [+ A1]) @ W + bias), W given as WT[n][k] bf16.
// Block: 16 output rows; 8 waves, one 16-col tile each; K=128 -> 4 WMMAs.
// Epilogue repacks C through LDS so global stores are b128 row-major.
// ---------------------------------------------------------------------------
template<bool HAS_A1, bool HAS_OUTB>
__global__ __launch_bounds__(256) void gemm_relu_kernel(
    const float* __restrict__ A0, const float* __restrict__ A1,
    const __bf16* __restrict__ WT, const float* __restrict__ bias,
    float* __restrict__ outF, __bf16* __restrict__ outB)
{
  __shared__ __bf16 lds_a[16][136];           // 136 = 128 + 8 pad (bank spread)
  __shared__ float  lds_o[16][132];           // output repack tile
  const int tid  = threadIdx.x;
  const int tile = blockIdx.x;
  const int row  = tid >> 4, part = tid & 15; // 16 threads per row, 8 floats each
  {
    size_t gi = (size_t)(tile * 16 + row) * HID + part * 8;
    float4 f0 = *(const float4*)(A0 + gi);
    float4 f1 = *(const float4*)(A0 + gi + 4);
    if (HAS_A1) {
      float4 m0 = *(const float4*)(A1 + gi);
      float4 m1 = *(const float4*)(A1 + gi + 4);
      f0.x += m0.x; f0.y += m0.y; f0.z += m0.z; f0.w += m0.w;
      f1.x += m1.x; f1.y += m1.y; f1.z += m1.z; f1.w += m1.w;
    }
    *(v8bf*)&lds_a[row][part * 8] = pack8(f0, f1);
  }
  __syncthreads();

  const int ct = tid >> 5, lane = tid & 31;
  const int lh = lane >> 4, m = lane & 15;    // half-wave, row-in-tile
  const int gn = ct * 16 + m;                 // output column
  v8f acc = {};
  #pragma unroll
  for (int kk = 0; kk < 4; ++kk) {
    // A slots: k = kk*32 + 16*half + i -> 16 contiguous bf16 of lds_a row m
    const __bf16* ar = &lds_a[m][kk * 32 + 16 * lh];
    v16bf a = cat8(*(const v8bf*)ar, *(const v8bf*)(ar + 8));
    // B slots: k = kk*32 + 16*half + i -> 16 contiguous bf16 of WT row gn
    const __bf16* br = WT + (size_t)gn * HID + kk * 32 + 16 * lh;
    v16bf b = cat8(*(const v8bf*)br, *(const v8bf*)(br + 8));
    acc = __builtin_amdgcn_wmma_f32_16x16x32_bf16(false, a, false, b, (short)0, acc, false, false);
  }
  float bv = bias[gn];
  #pragma unroll
  for (int r = 0; r < 8; ++r)                 // C/D layout: M = r + 8*half
    lds_o[r + 8 * lh][gn] = fmaxf(acc[r] + bv, 0.0f);
  __syncthreads();
  {
    float4 f0 = *(const float4*)&lds_o[row][part * 8];
    float4 f1 = *(const float4*)&lds_o[row][part * 8 + 4];
    size_t oi = (size_t)(tile * 16 + row) * HID + part * 8;
    *(float4*)(outF + oi)     = f0;
    *(float4*)(outF + oi + 4) = f1;
    if (HAS_OUTB) *(v8bf*)(outB + oi) = pack8(f0, f1);
  }
}

// ---------------------------------------------------------------------------
// Edge-mask MLP: 16 edges/block. ef=[h[src],h[dst]] (bf16, LDS), K=256 -> 8 WMMAs
// per wave; relu, dot with W_m2, cross-lane + cross-wave reduce, double sigmoid.
// ---------------------------------------------------------------------------
__global__ __launch_bounds__(256) void edge_mask_kernel(
    const __bf16* __restrict__ hB, const int* __restrict__ src, const int* __restrict__ dst,
    const __bf16* __restrict__ WTm1, const float* __restrict__ bm1,
    const float* __restrict__ Wm2, const float* __restrict__ bm2,
    float* __restrict__ maskOut)
{
  __shared__ __bf16 ef[16][264];              // 264 = 256 + 8 pad
  __shared__ float  esum[16];
  const int tid  = threadIdx.x;
  const int tile = blockIdx.x;
  if (tid < 16) esum[tid] = 0.0f;
  {
    int le = tid >> 4, part = tid & 15;
    int e = tile * 16 + le;
    int s = src[e], d = dst[e];
    v8bf vs = *(const v8bf*)(hB + (size_t)s * HID + part * 8);
    v8bf vd = *(const v8bf*)(hB + (size_t)d * HID + part * 8);
    *(v8bf*)&ef[le][part * 8]       = vs;
    *(v8bf*)&ef[le][HID + part * 8] = vd;
  }
  __syncthreads();

  const int ct = tid >> 5, lane = tid & 31;
  const int lh = lane >> 4, m = lane & 15;
  const int gn = ct * 16 + m;                 // hidden unit
  v8f acc = {};
  #pragma unroll
  for (int kk = 0; kk < 8; ++kk) {
    const __bf16* ar = &ef[m][kk * 32 + 16 * lh];
    v16bf a = cat8(*(const v8bf*)ar, *(const v8bf*)(ar + 8));
    const __bf16* br = WTm1 + (size_t)gn * 256 + kk * 32 + 16 * lh;
    v16bf b = cat8(*(const v8bf*)br, *(const v8bf*)(br + 8));
    acc = __builtin_amdgcn_wmma_f32_16x16x32_bf16(false, a, false, b, (short)0, acc, false, false);
  }
  float bv = bm1[gn], wv = Wm2[gn];
  #pragma unroll
  for (int r = 0; r < 8; ++r) {
    float v = fmaxf(acc[r] + bv, 0.0f) * wv;  // relu + W_m2 weight
    v += __shfl_xor(v, 1, 16);                // reduce over the 16 N-lanes
    v += __shfl_xor(v, 2, 16);
    v += __shfl_xor(v, 4, 16);
    v += __shfl_xor(v, 8, 16);
    if (m == 0) atomicAdd(&esum[r + 8 * lh], v);
  }
  __syncthreads();
  if (tid < 16) {
    float imp = 1.0f / (1.0f + __expf(-(esum[tid] + bm2[0])));
    float g   = (imp - 0.4f) * 2.0f;
    maskOut[tile * 16 + tid] = 1.0f / (1.0f + __expf(-g));
  }
}

// ---------------------------------------------------------------------------
// msgs[dst] += h[src] * mask[e] : one wave per edge, 4 floats/lane, L2 atomics.
// ---------------------------------------------------------------------------
__global__ __launch_bounds__(256) void scatter_kernel(
    const float* __restrict__ h, const float* __restrict__ mask,
    const int* __restrict__ src, const int* __restrict__ dst,
    float* __restrict__ msgs)
{
  int wave = threadIdx.x >> 5, lane = threadIdx.x & 31;
  int e = blockIdx.x * 8 + wave;
  int s = src[e], d = dst[e];
  float mk = mask[e];
  float4 hv = *(const float4*)(h + (size_t)s * HID + lane * 4);
  float* mp = msgs + (size_t)d * HID + lane * 4;
  unsafeAtomicAdd(mp + 0, hv.x * mk);
  unsafeAtomicAdd(mp + 1, hv.y * mk);
  unsafeAtomicAdd(mp + 2, hv.z * mk);
  unsafeAtomicAdd(mp + 3, hv.w * mk);
}

// ---------------------------------------------------------------------------
// prediction = relu(h[0] @ W_p1 + b_p1) @ W_p2 + b_p2
// ---------------------------------------------------------------------------
__global__ __launch_bounds__(128) void predict_kernel(
    const float* __restrict__ h, const float* __restrict__ Wp1, const float* __restrict__ bp1,
    const float* __restrict__ Wp2, const float* __restrict__ bp2, float* __restrict__ out)
{
  __shared__ float red[128];
  int j = threadIdx.x;
  float s = bp1[j];
  for (int k = 0; k < HID; ++k) s += h[k] * Wp1[k * HID + j];
  s = fmaxf(s, 0.0f) * Wp2[j];
  red[j] = s; __syncthreads();
  for (int off = 64; off > 0; off >>= 1) {
    if (j < off) red[j] += red[j + off];
    __syncthreads();
  }
  if (j == 0) out[0] = red[0] + bp2[0];
}

// ---------------------------------------------------------------------------
extern "C" void kernel_launch(void* const* d_in, const int* in_sizes, int n_in,
                              void* d_out, int out_size, void* d_ws, size_t ws_size,
                              hipStream_t stream) {
  const float* X     = (const float*)d_in[0];
  const int*   ei    = (const int*)d_in[1];
  const int*   src   = ei;
  const int*   dst   = ei + N_EDGES;
  const float* W_emb = (const float*)d_in[2];
  const float* b_emb = (const float*)d_in[3];
  const float* W_gnn = (const float*)d_in[4];
  const float* b_gnn = (const float*)d_in[5];
  const float* W_m1  = (const float*)d_in[6];
  const float* b_m1  = (const float*)d_in[7];
  const float* W_m2  = (const float*)d_in[8];
  const float* b_m2  = (const float*)d_in[9];
  const float* W_p1  = (const float*)d_in[10];
  const float* b_p1  = (const float*)d_in[11];
  const float* W_p2  = (const float*)d_in[12];
  const float* b_p2  = (const float*)d_in[13];

  float* out     = (float*)d_out;
  float* maskOut = out + 1;              // reference returns (pred, mask, h)
  float* hOut    = out + 1 + N_EDGES;

  char* ws = (char*)d_ws;
  float*  hF   = (float*)ws;  ws += (size_t)N_NODES * HID * sizeof(float);
  float*  msgs = (float*)ws;  ws += (size_t)N_NODES * HID * sizeof(float);
  __bf16* hB   = (__bf16*)ws; ws += (size_t)N_NODES * HID * sizeof(__bf16);
  __bf16* WTemb = (__bf16*)ws; ws += 128 * 128 * sizeof(__bf16);
  __bf16* WTg0  = (__bf16*)ws; ws += 128 * 128 * sizeof(__bf16);
  __bf16* WTg1  = (__bf16*)ws; ws += 128 * 128 * sizeof(__bf16);
  __bf16* WTg2  = (__bf16*)ws; ws += 128 * 128 * sizeof(__bf16);
  __bf16* WTm1  = (__bf16*)ws; ws += 256 * 128 * sizeof(__bf16);
  __bf16* WTg[3] = {WTg0, WTg1, WTg2};

  // weight prep (cheap, deterministic each call)
  transpose_to_bf16<<<(128 * 128) / 256, 256, 0, stream>>>(W_emb, WTemb, 128, 128);
  for (int i = 0; i < 3; ++i)
    transpose_to_bf16<<<(128 * 128) / 256, 256, 0, stream>>>(W_gnn + (size_t)i * 128 * 128, WTg[i], 128, 128);
  transpose_to_bf16<<<(256 * 128) / 256, 256, 0, stream>>>(W_m1, WTm1, 256, 128);

  // h = relu(X @ W_emb + b)
  gemm_relu_kernel<false, true><<<N_NODES / 16, 256, 0, stream>>>(X, nullptr, WTemb, b_emb, hF, hB);
  // mask0
  edge_mask_kernel<<<N_EDGES / 16, 256, 0, stream>>>(hB, src, dst, WTm1, b_m1, W_m2, b_m2, maskOut);

  for (int i = 0; i < 3; ++i) {
    zero_kernel<<<(N_NODES * HID) / (256 * 4), 256, 0, stream>>>(msgs, N_NODES * HID);
    scatter_kernel<<<N_EDGES / 8, 256, 0, stream>>>(hF, maskOut, src, dst, msgs);
    if (i < 2) {
      gemm_relu_kernel<true, true><<<N_NODES / 16, 256, 0, stream>>>(
          hF, msgs, WTg[i], b_gnn + (size_t)i * HID, hF, hB);
      edge_mask_kernel<<<N_EDGES / 16, 256, 0, stream>>>(hB, src, dst, WTm1, b_m1, W_m2, b_m2, maskOut);
    } else {
      gemm_relu_kernel<true, false><<<N_NODES / 16, 256, 0, stream>>>(
          hF, msgs, WTg[i], b_gnn + (size_t)i * HID, hOut, nullptr);
    }
  }

  predict_kernel<<<1, 128, 0, stream>>>(hOut, W_p1, b_p1, W_p2, b_p2, out);
}
